// CosineSimHashDecoder_74105365725422
// MI455X (gfx1250) — matrix operand order repaired and checked by reading, not compiled, not run on hardware
//
#include <hip/hip_runtime.h>
#include <hip/hip_bf16.h>
#include <stdint.h>

// ---------------------------------------------------------------------------
// CosineSimHashDecoder on MI455X (gfx1250, wave32)
//
// Roofline: output = 8192*8192*4B = 256 MiB -> ~11us floor @ 23.3 TB/s HBM.
// Gram GEMM = 34.4 GFLOP; with v_wmma_f32_16x16x32_f16 this is ~ the memory
// floor, while f32 WMMA (16x16x4) would dominate. zn (4 MiB f16) lives in L2
// (192 MB), so fragment re-reads are L2 hits; kernel is store-bound.
// ---------------------------------------------------------------------------

#define N_NODES 8192
#define DIM     256
#define NBANDS  64            // 64 bands * 8 bits = 512 projections
#define NPROJ   512

typedef __attribute__((ext_vector_type(16))) _Float16 v16h;
typedef __attribute__((ext_vector_type(8)))  _Float16 v8h;
typedef __attribute__((ext_vector_type(8)))  float    v8f;

union Frag16 { v16h v; v8h h[2]; };

// ---------------------------------------------------------------------------
// Kernel 1: row-normalize z (f32) -> zn (f16). One block per row.
// ---------------------------------------------------------------------------
__global__ __launch_bounds__(256) void normalize_kernel(
    const float* __restrict__ z, _Float16* __restrict__ zn)
{
    __shared__ float red[256];
    const int i = blockIdx.x;
    const int t = threadIdx.x;
    const float v = z[(size_t)i * DIM + t];
    red[t] = v * v;
    __syncthreads();
    #pragma unroll
    for (int s = 128; s > 0; s >>= 1) {
        if (t < s) red[t] += red[t + s];
        __syncthreads();
    }
    const float rn = rsqrtf(red[0]);
    zn[(size_t)i * DIM + t] = (_Float16)(v * rn);
}

// ---------------------------------------------------------------------------
// Kernel 2: SimHash bucket codes. One block per row: each thread computes two
// of the 512 projections (sign of z . plane), then 64 threads pack 8 sign
// bits -> one bucket byte per band. 2.1 GFLOP total: negligible.
// ---------------------------------------------------------------------------
__global__ __launch_bounds__(256) void codes_kernel(
    const float* __restrict__ z, const float* __restrict__ planes,
    unsigned char* __restrict__ codes)
{
    __shared__ float zrow[DIM];
    __shared__ unsigned char bits[NPROJ];
    const int i = blockIdx.x;
    const int t = threadIdx.x;
    zrow[t] = z[(size_t)i * DIM + t];
    __syncthreads();
    float a0 = 0.0f, a1 = 0.0f;
    #pragma unroll 8
    for (int k = 0; k < DIM; ++k) {
        const float zk = zrow[k];
        a0 = fmaf(zk, planes[(size_t)k * NPROJ + t],        a0);
        a1 = fmaf(zk, planes[(size_t)k * NPROJ + t + 256],  a1);
    }
    bits[t]       = (a0 >= 0.0f) ? 1u : 0u;
    bits[t + 256] = (a1 >= 0.0f) ? 1u : 0u;
    __syncthreads();
    if (t < NBANDS) {
        unsigned c = 0;
        #pragma unroll
        for (int r = 0; r < 8; ++r) c |= (unsigned)bits[t * 8 + r] << r;
        codes[(size_t)i * NBANDS + t] = (unsigned char)c;
    }
}

// ---------------------------------------------------------------------------
// Kernel 3: Gram GEMM via v_wmma_f32_16x16x32_f16 + LSH filter + store.
// Grid 64x64 blocks; each block = 128x128 output tile, 8 waves (wave32),
// each wave = 64x32 sub-tile = 4x2 WMMA accumulators, K-loop 256/32 = 8.
//
// Symmetry trick: B = zn^T, so the B fragment for output columns is loaded
// from ROWS of zn, just like A — both as 16B vector loads matching the
// documented 16-bit A/B VGPR layouts (A lane-half: K {0..7}|{16..23};
// B lane-half: same K as A).
// ---------------------------------------------------------------------------
__global__ __launch_bounds__(256) void simhash_gemm_kernel(
    const _Float16* __restrict__ zn,
    const uint64_t* __restrict__ codes64,   // [N_NODES][8] u64 (64 bands)
    float* __restrict__ out)
{
    __shared__ uint64_t rowC[128 * 8];
    __shared__ uint64_t colC[128 * 8];

    const int bn = blockIdx.x;   // column tile
    const int bm = blockIdx.y;   // row tile
    const int tid = threadIdx.x;

    // Stage the 64-byte bucket codes for this tile's 128 rows + 128 cols.
    for (int idx = tid; idx < 128 * 8; idx += 256) {
        const int r = idx >> 3, w = idx & 7;
        rowC[idx] = codes64[(size_t)(bm * 128 + r) * 8 + w];
        colC[idx] = codes64[(size_t)(bn * 128 + r) * 8 + w];
    }
    __syncthreads();

    const int wave = tid >> 5;
    const int lane = tid & 31;
    const int wm = wave >> 2;          // 0..1  (row group of 64)
    const int wn = wave & 3;           // 0..3  (col group of 32)
    const int lm = lane & 15;          // M/N within 16x16 tile
    const int lh = lane >> 4;          // lane half -> K-half select

    const int r0 = bm * 128 + wm * 64; // global row base of this wave
    const int c0 = bn * 128 + wn * 32; // global col base of this wave

    v8f acc[4][2] = {};

    #pragma unroll
    for (int kk = 0; kk < DIM; kk += 32) {
        Frag16 A[4], B[2];
        // A fragments: lane L, row r0+16m+lm; K = kk + lh*8 + {0..7, 16..23}
        #pragma unroll
        for (int m = 0; m < 4; ++m) {
            const _Float16* base =
                zn + (size_t)(r0 + m * 16 + lm) * DIM + kk + lh * 8;
            A[m].h[0] = *(const v8h*)(base);        // K +0..7
            A[m].h[1] = *(const v8h*)(base + 16);   // K +16..23
        }
        // B fragments (B = zn^T): lane L, col c0+16n+lm; K = kk + lh*8 + {0..7, 16..23}
        #pragma unroll
        for (int n = 0; n < 2; ++n) {
            const _Float16* base =
                zn + (size_t)(c0 + n * 16 + lm) * DIM + kk + lh * 8;
            B[n].h[0] = *(const v8h*)(base);        // K +0..7
            B[n].h[1] = *(const v8h*)(base + 16);   // K +16..23
        }
        #pragma unroll
        for (int m = 0; m < 4; ++m)
            #pragma unroll
            for (int n = 0; n < 2; ++n)
                acc[m][n] = __builtin_amdgcn_wmma_f32_16x16x32_f16(
                    /*neg_a=*/false, A[m].v, /*neg_b=*/false, B[n].v,
                    /*c_mod=*/(short)0, acc[m][n],
                    /*reuse_a=*/false, /*reuse_b=*/false);
    }

    // Epilogue. C/D layout: VGPR v -> M = v + 8*(lane>>4), N = lane&15.
    // cos >= 0.75 (dist <= 0.25) virtually never fires for random Gaussian
    // pairs, so the band compare is behind a cold predicate.
    #pragma unroll
    for (int m = 0; m < 4; ++m) {
        #pragma unroll
        for (int n = 0; n < 2; ++n) {
            #pragma unroll
            for (int v = 0; v < 8; ++v) {
                const int lr   = wm * 64 + m * 16 + v + lh * 8; // local row
                const int lc   = wn * 32 + n * 16 + lm;         // local col
                const int grow = bm * 128 + lr;
                const int gcol = bn * 128 + lc;
                const float cosv = acc[m][n][v];
                float o = 0.0f;
                if (grow == gcol) {
                    o = 1.0f;                      // identity diagonal
                } else if (cosv >= 0.75f) {
                    // any of 64 bucket bytes equal? 8 x (u64 XOR + has-zero-byte)
                    bool hit = false;
                    #pragma unroll
                    for (int b = 0; b < 8; ++b) {
                        const uint64_t x = rowC[lr * 8 + b] ^ colC[lc * 8 + b];
                        hit |= (((x - 0x0101010101010101ull) & ~x) &
                                 0x8080808080808080ull) != 0ull;
                    }
                    if (hit) o = cosv;             // 1 - dist == cos
                }
                out[(size_t)grow * N_NODES + gcol] = o;
            }
        }
    }
}

// ---------------------------------------------------------------------------
// Launch: normalize -> codes -> tiled WMMA GEMM+filter, all on `stream`.
// Workspace layout: [0, 4MiB) zn f16; [4MiB, 4.5MiB) codes u8.
// ---------------------------------------------------------------------------
extern "C" void kernel_launch(void* const* d_in, const int* in_sizes, int n_in,
                              void* d_out, int out_size, void* d_ws, size_t ws_size,
                              hipStream_t stream) {
    const float* z      = (const float*)d_in[0];   // [8192, 256] f32
    const float* planes = (const float*)d_in[1];   // [256, 512]  f32
    float*       out    = (float*)d_out;           // [8192, 8192] f32

    _Float16*      zn    = (_Float16*)d_ws;
    unsigned char* codes = (unsigned char*)d_ws +
                           (size_t)N_NODES * DIM * sizeof(_Float16);

    normalize_kernel<<<N_NODES, 256, 0, stream>>>(z, zn);
    codes_kernel<<<N_NODES, 256, 0, stream>>>(z, planes, codes);

    dim3 grid(N_NODES / 128, N_NODES / 128);       // 64 x 64 tiles
    simhash_gemm_kernel<<<grid, 256, 0, stream>>>(
        zn, (const uint64_t*)codes, out);
}
